// TopologyAugmentor_8014408974928
// MI455X (gfx1250) — matrix operand order repaired and checked by reading, not compiled, not run on hardware
//
#include <hip/hip_runtime.h>
#include <stdint.h>

// ---------------- problem constants ----------------
#define NN    8192
#define DD    128
#define EE    131072
#define KTOP  524288
#define HSZ   (1u << 18)
#define HMASK (HSZ - 1u)
#define NBINS 4096
#define HEMPTY 0xFFFFFFFFu

// output layout (flat f32 elements)
#define OUT_X     0
#define OUT_EI    1048576
#define OUT_ROWS  1310720
#define OUT_COLS  1835008
#define OUT_W     2359296

typedef __bf16 bf16;
typedef bf16  v8bf  __attribute__((ext_vector_type(8)));
typedef bf16  v16bf __attribute__((ext_vector_type(16)));
typedef float v8f   __attribute__((ext_vector_type(8)));

// ---------------- workspace layout (bytes, all 256B aligned) ----------------
constexpr size_t SZ_NF = (size_t)NN * DD * 4;     // 4 MiB f32 node-feature buf
constexpr size_t SZ_NH = (size_t)NN * DD * 2;     // 2 MiB bf16 node-feature buf
constexpr size_t O_DEG  = 0;
constexpr size_t O_DIS  = O_DEG  + (size_t)NN * 4;
constexpr size_t O_HW1  = O_DIS  + (size_t)NN * 4;
constexpr size_t O_AGG  = O_HW1  + SZ_NF;
constexpr size_t O_H1   = O_AGG  + SZ_NF;
constexpr size_t O_HW2  = O_H1   + SZ_NF;
constexpr size_t O_H2   = O_HW2  + SZ_NF;
constexpr size_t O_XN   = O_H2   + SZ_NF;
constexpr size_t O_XB   = O_XN   + SZ_NF;
constexpr size_t O_H1B  = O_XB   + SZ_NH;
constexpr size_t O_XNB  = O_H1B  + SZ_NH;
constexpr size_t O_W1T  = O_XNB  + SZ_NH;
constexpr size_t O_W2T  = O_W1T  + (size_t)DD * DD * 2;
constexpr size_t O_KEYS = O_W2T  + (size_t)DD * DD * 2;
constexpr size_t O_HCNT = O_KEYS + (size_t)HSZ * 4;
constexpr size_t O_HIST = O_HCNT + (size_t)HSZ * 4;
constexpr size_t O_SCAL = O_HIST + (size_t)NBINS * 4;
constexpr size_t O_RMAX = O_SCAL + 256;
constexpr size_t O_RSUM = O_RMAX + (size_t)NN * 4;
constexpr size_t O_ROWS = O_RSUM + (size_t)NN * 4;
constexpr size_t O_VALS = O_ROWS + (size_t)KTOP * 4;
// total ~= 36.2 MiB

// scal[]: 0=global max (ordered u32), 1=hi counter, 2=mid counter,
//         3=threshold bucket t, 4=count above t, 5=need from bucket t

// ---------------- helpers ----------------
__device__ __forceinline__ unsigned f2ord(float f) {
    unsigned u = __float_as_uint(f);
    return (u & 0x80000000u) ? ~u : (u | 0x80000000u);
}
__device__ __forceinline__ float ord2f(unsigned u) {
    u = (u & 0x80000000u) ? (u & 0x7FFFFFFFu) : ~u;
    return __uint_as_float(u);
}
__device__ __forceinline__ int bucketi(float v) {
    int b = (int)((v + 1.0f) * 1024.0f);
    return b < 0 ? 0 : (b > (NBINS - 1) ? (NBINS - 1) : b);
}

// ---- register-blocked WMMA: one wave computes a 32x64 output region ----
// A:  row-major [M x K] bf16.   Bt: row-major [N x K] bf16 (B transposed).
// acc[mi][ni] = 16x16 tile (m0+16*mi, n0+16*ni). Wave32; A/B k = 16*(lane/16) + s (contiguous), C as in ISA 7.12.2.
__device__ __forceinline__ void tileAcc(const bf16* __restrict__ A,
                                        const bf16* __restrict__ Bt,
                                        int m0, int n0, int K,
                                        v8f c[2][4]) {
    const int lane = threadIdx.x & 31;
    const int hf   = (lane >> 4) & 1;
    const int akb  = hf * 16;
    const int bkb  = hf * 16;
    const bf16* ar0 = A + (size_t)(m0 + (lane & 15)) * K + akb;
    const bf16* ar1 = ar0 + (size_t)16 * K;
    const bf16* br0 = Bt + (size_t)(n0 + (lane & 15)) * K + bkb;
#pragma unroll
    for (int mi = 0; mi < 2; ++mi)
#pragma unroll
        for (int ni = 0; ni < 4; ++ni) c[mi][ni] = (v8f){};
#pragma unroll
    for (int k0 = 0; k0 < K; k0 += 32) {
        v16bf a[2], b[4];
#pragma unroll
        for (int mi = 0; mi < 2; ++mi) {
            const bf16* ap = (mi ? ar1 : ar0) + k0;
            v8bf alo = *(const v8bf*)(ap);
            v8bf ahi = *(const v8bf*)(ap + 8);
            a[mi] = __builtin_shufflevector(alo, ahi,
                     0,1,2,3,4,5,6,7,8,9,10,11,12,13,14,15);
        }
#pragma unroll
        for (int ni = 0; ni < 4; ++ni)
            b[ni] = *(const v16bf*)(br0 + (size_t)ni * 16 * K + k0);
#pragma unroll
        for (int mi = 0; mi < 2; ++mi)
#pragma unroll
            for (int ni = 0; ni < 4; ++ni)
                c[mi][ni] = __builtin_amdgcn_wmma_f32_16x16x32_bf16(
                        false, a[mi], false, b[ni], (short)0, c[mi][ni],
                        false, false);
    }
}

// ---------------- kernels ----------------
__global__ __launch_bounds__(256) void k_init(float* deg, unsigned* keys,
        unsigned* hcnt, unsigned* hist, unsigned* scal, unsigned* rmax,
        float* rsum) {
    unsigned i = blockIdx.x * 256 + threadIdx.x;
    if (i < HSZ)  { keys[i] = HEMPTY; hcnt[i] = 0u; }
    if (i < NN)   { deg[i] = 1.0f; rmax[i] = 0u; rsum[i] = 0.0f; }
    if (i < NBINS) hist[i] = 0u;
    if (i < 16)    scal[i] = 0u;
}

__global__ __launch_bounds__(256) void k_zero_f32(float* p, int n) {
    int i = blockIdx.x * 256 + threadIdx.x;
    if (i < n) p[i] = 0.0f;
}

__global__ __launch_bounds__(256) void k_deg(const int* ei, float* deg) {
    int i = blockIdx.x * 256 + threadIdx.x;
    if (i < EE) atomicAdd(&deg[ei[EE + i]], 1.0f);
}

__global__ __launch_bounds__(256) void k_dis(const float* deg, float* dis) {
    int i = blockIdx.x * 256 + threadIdx.x;
    if (i < NN) dis[i] = rsqrtf(deg[i]);
}

__global__ __launch_bounds__(256) void k_cvt(const float* a, bf16* b, int n) {
    int i = blockIdx.x * 256 + threadIdx.x;
    if (i < n) b[i] = (bf16)a[i];
}

__global__ __launch_bounds__(256) void k_cvtT(const float* w, bf16* wt) {
    int i = blockIdx.x * 256 + threadIdx.x;
    if (i < DD * DD) {
        int n = i >> 7, k = i & 127;
        wt[n * DD + k] = (bf16)w[k * DD + n];
    }
}

// C[M x Nn] = A[M x K] * Bt[Nn x K]^T ; one 32x64 region per wave
__global__ __launch_bounds__(256) void k_gemm(float* __restrict__ C,
        const bf16* __restrict__ A, const bf16* __restrict__ Bt,
        int M, int Nn, int K) {
    int wave = (blockIdx.x * 256 + threadIdx.x) >> 5;
    int rn = Nn >> 6;                       // regions along N
    int tm = (wave / rn) << 5;
    int tn = (wave % rn) << 6;
    if (tm >= M) return;                    // wave-uniform
    v8f c[2][4];
    tileAcc(A, Bt, tm, tn, K, c);
    int lane = threadIdx.x & 31;
    int coll = lane & 15;
    int rb   = (lane & 16) ? 8 : 0;
#pragma unroll
    for (int mi = 0; mi < 2; ++mi)
#pragma unroll
        for (int ni = 0; ni < 4; ++ni)
#pragma unroll
            for (int v = 0; v < 8; ++v)
                C[(size_t)(tm + mi * 16 + rb + v) * Nn + tn + ni * 16 + coll]
                    = c[mi][ni][v];
}

__global__ __launch_bounds__(256) void k_scatter(float* __restrict__ agg,
        const float* __restrict__ hw, const int* __restrict__ ei,
        const float* __restrict__ dis) {
    int gid = blockIdx.x * 256 + threadIdx.x;
    if (gid >= EE * DD) return;
    int e = gid >> 7, f = gid & 127;
    int s = ei[e], d = ei[EE + e];
    atomicAdd(&agg[(size_t)d * DD + f], hw[(size_t)s * DD + f] * dis[s] * dis[d]);
}

__global__ __launch_bounds__(256) void k_h1(float* h1, bf16* h1b,
        const float* agg, const float* hw1, const float* dis) {
    int i = blockIdx.x * 256 + threadIdx.x;
    if (i >= NN * DD) return;
    float ds = dis[i >> 7];
    float v = agg[i] + hw1[i] * ds * ds;
    v = fmaxf(v, 0.0f);
    h1[i] = v; h1b[i] = (bf16)v;
}

__global__ __launch_bounds__(256) void k_h2(float* h2, const float* agg,
        const float* hw2, const float* h1, const float* dis) {
    int i = blockIdx.x * 256 + threadIdx.x;
    if (i >= NN * DD) return;
    float ds = dis[i >> 7];
    h2[i] = agg[i] + hw2[i] * ds * ds + h1[i];
}

__global__ __launch_bounds__(256) void k_rownorm(const float* h2, float* xn,
                                                 bf16* xnb) {
    int r = blockIdx.x * 256 + threadIdx.x;
    if (r >= NN) return;
    const float* p = h2 + (size_t)r * DD;
    float s = 0.0f;
#pragma unroll 4
    for (int i = 0; i < DD; ++i) s += p[i] * p[i];
    float inv = 1.0f / (sqrtf(s) + 1e-12f);
#pragma unroll 4
    for (int i = 0; i < DD; ++i) {
        float v = p[i] * inv;
        xn[(size_t)r * DD + i]  = v;
        xnb[(size_t)r * DD + i] = (bf16)v;
    }
}

__global__ __launch_bounds__(256) void k_hash(const int* ei, unsigned* keys,
                                              unsigned* hcnt) {
    int e = blockIdx.x * 256 + threadIdx.x;
    if (e >= EE) return;
    unsigned key = (unsigned)ei[e] * (unsigned)NN + (unsigned)ei[EE + e];
    unsigned h = (key * 2654435761u) & HMASK;
    for (;;) {
        unsigned old = atomicCAS(&keys[h], HEMPTY, key);
        if (old == HEMPTY || old == key) { atomicAdd(&hcnt[h], 1u); break; }
        h = (h + 1u) & HMASK;
    }
}

// pass 1: histogram of un-boosted adj + global max (LDS-accumulated)
// one 32x64 region per wave; 4096 blocks total
__global__ __launch_bounds__(256) void k_adj1(const bf16* __restrict__ xnb,
        unsigned* __restrict__ hist, unsigned* __restrict__ scal) {
    __shared__ unsigned lhist[NBINS];
    __shared__ unsigned lmax;
    int tid = threadIdx.x;
    for (int i = tid; i < NBINS; i += 256) lhist[i] = 0u;
    if (tid == 0) lmax = 0u;
    __syncthreads();
    int w  = blockIdx.x * 8 + (tid >> 5);
    int tm = (w >> 7) << 5;                 // 256 regions along rows
    int tn = (w & 127) << 6;                // 128 regions along cols
    v8f c[2][4];
    tileAcc(xnb, xnb, tm, tn, DD, c);
    float mx = c[0][0][0];
#pragma unroll
    for (int mi = 0; mi < 2; ++mi)
#pragma unroll
        for (int ni = 0; ni < 4; ++ni)
#pragma unroll
            for (int v = 0; v < 8; ++v) {
                float cv = c[mi][ni][v];
                mx = fmaxf(mx, cv);
                atomicAdd(&lhist[bucketi(cv)], 1u);
            }
    atomicMax(&lmax, f2ord(mx));
    __syncthreads();
    for (int i = tid; i < NBINS; i += 256) {
        unsigned h = lhist[i];
        if (h) atomicAdd(&hist[i], h);
    }
    if (tid == 0) atomicMax(&scal[0], lmax);
}

// histogram fix-up for the boosted (original-edge) entries
__global__ __launch_bounds__(256) void k_fix(const unsigned* keys,
        const unsigned* hcnt, const float* __restrict__ xn,
        unsigned* hist, const unsigned* scal) {
    unsigned s = blockIdx.x * 256 + threadIdx.x;
    if (s >= HSZ) return;
    unsigned k = keys[s];
    if (k == HEMPTY) return;
    unsigned r = k >> 13, cc = k & 8191u;
    const float* a = xn + (size_t)r  * DD;
    const float* b = xn + (size_t)cc * DD;
    float d = 0.0f;
#pragma unroll 4
    for (int i = 0; i < DD; ++i) d += a[i] * b[i];
    float g = ord2f(scal[0]);
    atomicSub(&hist[bucketi(d)], 1u);
    atomicAdd(&hist[bucketi(d + (float)hcnt[s] * g)], 1u);
}

__global__ void k_scan(const unsigned* hist, unsigned* scal) {
    if (threadIdx.x != 0 || blockIdx.x != 0) return;
    unsigned acc = 0;
    for (int b = NBINS - 1; b >= 0; --b) {
        int hv = (int)hist[b];              // clamp fix-up underflow
        unsigned h = hv < 0 ? 0u : (unsigned)hv;
        if (acc + h >= (unsigned)KTOP) {
            scal[3] = (unsigned)b;
            scal[4] = acc;
            scal[5] = (unsigned)KTOP - acc;
            return;
        }
        acc += h;
    }
    scal[3] = 0; scal[4] = acc; scal[5] = (unsigned)KTOP - acc;
}

// pass 2: recompute regions; emit non-boosted entries above threshold.
// Boost only raises values, so anything below bucket t that is boosted is
// handled by k_edgeemit; here we probe the hash ONLY for rare candidates
// (bucket >= t) to exclude boosted duplicates.
__global__ __launch_bounds__(256) void k_adj2(const bf16* __restrict__ xnb,
        const unsigned* __restrict__ keys, unsigned* scal,
        int* __restrict__ rows, float* __restrict__ vals,
        float* __restrict__ outR, float* __restrict__ outC) {
    int tid = threadIdx.x;
    int w  = blockIdx.x * 8 + (tid >> 5);
    int tm = (w >> 7) << 5;
    int tn = (w & 127) << 6;
    v8f c[2][4];
    tileAcc(xnb, xnb, tm, tn, DD, c);
    int lane = tid & 31;
    int coll = lane & 15;
    int rbo  = (lane & 16) ? 8 : 0;
    int t        = (int)scal[3];
    unsigned abv = scal[4];
    unsigned nee = scal[5];
#pragma unroll
    for (int mi = 0; mi < 2; ++mi)
#pragma unroll
        for (int ni = 0; ni < 4; ++ni)
#pragma unroll
            for (int v = 0; v < 8; ++v) {
                float cv = c[mi][ni][v];
                int b = bucketi(cv);
                if (b < t) continue;
                int row = tm + mi * 16 + rbo + v;
                int col = tn + ni * 16 + coll;
                // exclude boosted entries (emitted by k_edgeemit)
                unsigned key = (unsigned)row * (unsigned)NN + (unsigned)col;
                unsigned h = (key * 2654435761u) & HMASK;
                bool boosted = false;
                for (;;) {
                    unsigned k = keys[h];
                    if (k == HEMPTY) break;
                    if (k == key) { boosted = true; break; }
                    h = (h + 1u) & HMASK;
                }
                if (boosted) continue;
                if (b > t) {
                    unsigned pos = atomicAdd(&scal[1], 1u);
                    if (pos < (unsigned)KTOP) {
                        rows[pos] = row; vals[pos] = cv;
                        outR[pos] = (float)row; outC[pos] = (float)col;
                    }
                } else {
                    unsigned q = atomicAdd(&scal[2], 1u);
                    if (q < nee) {
                        unsigned pos = abv + q;
                        rows[pos] = row; vals[pos] = cv;
                        outR[pos] = (float)row; outC[pos] = (float)col;
                    }
                }
            }
}

// emit the boosted (original-edge) entries from the hash table
__global__ __launch_bounds__(256) void k_edgeemit(const unsigned* keys,
        const unsigned* hcnt, const float* __restrict__ xn, unsigned* scal,
        int* __restrict__ rows, float* __restrict__ vals,
        float* __restrict__ outR, float* __restrict__ outC) {
    unsigned s = blockIdx.x * 256 + threadIdx.x;
    if (s >= HSZ) return;
    unsigned k = keys[s];
    if (k == HEMPTY) return;
    int row = (int)(k >> 13), col = (int)(k & 8191u);
    const float* a = xn + (size_t)row * DD;
    const float* bp = xn + (size_t)col * DD;
    float d = 0.0f;
#pragma unroll 4
    for (int i = 0; i < DD; ++i) d += a[i] * bp[i];
    float vf = d + (float)hcnt[s] * ord2f(scal[0]);
    int b = bucketi(vf);
    int t = (int)scal[3];
    if (b < t) return;
    unsigned abv = scal[4], nee = scal[5];
    if (b > t) {
        unsigned pos = atomicAdd(&scal[1], 1u);
        if (pos < (unsigned)KTOP) {
            rows[pos] = row; vals[pos] = vf;
            outR[pos] = (float)row; outC[pos] = (float)col;
        }
    } else {
        unsigned q = atomicAdd(&scal[2], 1u);
        if (q < nee) {
            unsigned pos = abv + q;
            rows[pos] = row; vals[pos] = vf;
            outR[pos] = (float)row; outC[pos] = (float)col;
        }
    }
}

__global__ __launch_bounds__(256) void k_smax(const int* rows,
        const float* vals, unsigned* rmax) {
    int i = blockIdx.x * 256 + threadIdx.x;
    if (i < KTOP) atomicMax(&rmax[rows[i]], f2ord(vals[i]));
}

__global__ __launch_bounds__(256) void k_ssum(const int* rows, float* vals,
        const unsigned* rmax, float* rsum) {
    int i = blockIdx.x * 256 + threadIdx.x;
    if (i >= KTOP) return;
    int r = rows[i];
    float e = __expf(vals[i] - ord2f(rmax[r]));
    vals[i] = e;
    atomicAdd(&rsum[r], e);
}

__global__ __launch_bounds__(256) void k_snorm(const int* rows,
        const float* vals, const float* rsum, float* outW) {
    int i = blockIdx.x * 256 + threadIdx.x;
    if (i < KTOP) outW[i] = vals[i] / rsum[rows[i]];
}

__global__ __launch_bounds__(256) void k_copy(const float* x, const int* ei,
                                              float* out) {
    int i = blockIdx.x * 256 + threadIdx.x;
    if (i < OUT_EI) out[i] = x[i];
    else if (i < OUT_ROWS) out[i] = (float)ei[i - OUT_EI];
}

// ---------------- launch ----------------
extern "C" void kernel_launch(void* const* d_in, const int* in_sizes, int n_in,
                              void* d_out, int out_size, void* d_ws,
                              size_t ws_size, hipStream_t stream) {
    const float* x  = (const float*)d_in[0];
    const int*   ei = (const int*)d_in[1];
    const float* W1 = (const float*)d_in[2];
    const float* W2 = (const float*)d_in[3];
    float* out = (float*)d_out;
    char* ws = (char*)d_ws;

    float*    deg  = (float*)(ws + O_DEG);
    float*    dis  = (float*)(ws + O_DIS);
    float*    hw1  = (float*)(ws + O_HW1);
    float*    agg  = (float*)(ws + O_AGG);
    float*    h1   = (float*)(ws + O_H1);
    float*    hw2  = (float*)(ws + O_HW2);
    float*    h2   = (float*)(ws + O_H2);
    float*    xn   = (float*)(ws + O_XN);
    bf16*     xb   = (bf16*)(ws + O_XB);
    bf16*     h1b  = (bf16*)(ws + O_H1B);
    bf16*     xnb  = (bf16*)(ws + O_XNB);
    bf16*     w1t  = (bf16*)(ws + O_W1T);
    bf16*     w2t  = (bf16*)(ws + O_W2T);
    unsigned* keys = (unsigned*)(ws + O_KEYS);
    unsigned* hcnt = (unsigned*)(ws + O_HCNT);
    unsigned* hist = (unsigned*)(ws + O_HIST);
    unsigned* scal = (unsigned*)(ws + O_SCAL);
    unsigned* rmax = (unsigned*)(ws + O_RMAX);
    float*    rsum = (float*)(ws + O_RSUM);
    int*      rows = (int*)(ws + O_ROWS);
    float*    vals = (float*)(ws + O_VALS);

    const int NF = NN * DD;

    k_init<<<HSZ / 256, 256, 0, stream>>>(deg, keys, hcnt, hist, scal, rmax, rsum);
    k_zero_f32<<<NF / 256, 256, 0, stream>>>(agg, NF);

    // degrees / normalization
    k_deg<<<EE / 256, 256, 0, stream>>>(ei, deg);
    k_dis<<<NN / 256, 256, 0, stream>>>(deg, dis);

    // bf16 operands
    k_cvt<<<NF / 256, 256, 0, stream>>>(x, xb, NF);
    k_cvtT<<<(DD * DD) / 256, 256, 0, stream>>>(W1, w1t);
    k_cvtT<<<(DD * DD) / 256, 256, 0, stream>>>(W2, w2t);

    // layer 1 (GEMM waves: (M/32)*(N/64))
    const int gemmBlocks = (NN / 32) * (DD / 64) * 32 / 256;   // 64
    k_gemm<<<gemmBlocks, 256, 0, stream>>>(hw1, xb, w1t, NN, DD, DD);
    k_scatter<<<(EE * DD) / 256, 256, 0, stream>>>(agg, hw1, ei, dis);
    k_h1<<<NF / 256, 256, 0, stream>>>(h1, h1b, agg, hw1, dis);

    // layer 2 + residual
    k_zero_f32<<<NF / 256, 256, 0, stream>>>(agg, NF);
    k_gemm<<<gemmBlocks, 256, 0, stream>>>(hw2, h1b, w2t, NN, DD, DD);
    k_scatter<<<(EE * DD) / 256, 256, 0, stream>>>(agg, hw2, ei, dis);
    k_h2<<<NF / 256, 256, 0, stream>>>(h2, agg, hw2, h1, dis);

    // cosine normalization
    k_rownorm<<<NN / 256, 256, 0, stream>>>(h2, xn, xnb);

    // original-edge hash (boost bookkeeping)
    k_hash<<<EE / 256, 256, 0, stream>>>(ei, keys, hcnt);

    // fused similarity passes (no 256MB adj materialization)
    const int adjBlocks = (NN / 32) * (NN / 64) / 8;   // 4096
    k_adj1<<<adjBlocks, 256, 0, stream>>>(xnb, hist, scal);
    k_fix<<<HSZ / 256, 256, 0, stream>>>(keys, hcnt, xn, hist, scal);
    k_scan<<<1, 32, 0, stream>>>(hist, scal);
    k_adj2<<<adjBlocks, 256, 0, stream>>>(xnb, keys, scal, rows, vals,
                                          out + OUT_ROWS, out + OUT_COLS);
    k_edgeemit<<<HSZ / 256, 256, 0, stream>>>(keys, hcnt, xn, scal, rows, vals,
                                              out + OUT_ROWS, out + OUT_COLS);

    // scatter softmax over rows
    k_smax<<<KTOP / 256, 256, 0, stream>>>(rows, vals, rmax);
    k_ssum<<<KTOP / 256, 256, 0, stream>>>(rows, vals, rmax, rsum);
    k_snorm<<<KTOP / 256, 256, 0, stream>>>(rows, vals, rsum, out + OUT_W);

    // passthrough outputs
    k_copy<<<OUT_ROWS / 256, 256, 0, stream>>>(x, ei, out);
}
